// MHSEAttn_76227079569421
// MI455X (gfx1250) — matrix operand order repaired and checked by reading, not compiled, not run on hardware
//
#include <hip/hip_runtime.h>
#include <hip/hip_bf16.h>
#include <cstdint>

// Problem constants (match reference)
#define BB   8
#define TT   1024
#define DD   1024
#define HH   16
#define CC   8
#define DKK  64
#define LAMBDV 5.0f

typedef __attribute__((ext_vector_type(16))) __bf16 v16bf;
typedef __attribute__((ext_vector_type(8)))  float  v8f;
typedef __attribute__((ext_vector_type(4)))  int    v4i;

#if __has_builtin(__builtin_amdgcn_global_load_async_to_lds_b128) && \
    __has_builtin(__builtin_amdgcn_s_wait_asynccnt)
#define USE_ASYNC_LDS 1
typedef v4i __attribute__((address_space(1))) as1_v4i;  // global b128 payload
typedef v4i __attribute__((address_space(3))) as3_v4i;  // LDS b128 payload
#else
#define USE_ASYNC_LDS 0
#endif

struct alignas(32) Frag32 { uint4 lo, hi; };

// native RNE f32 -> bf16 (lets clang emit v_cvt_*bf16_f32 instead of ALU chains)
__device__ __forceinline__ unsigned short f2bf(float f) {
    __bf16 h = (__bf16)f;
    return __builtin_bit_cast(unsigned short, h);
}

__device__ __forceinline__ v16bf load_frag(const unsigned short* p0,
                                           const unsigned short* p1) {
    Frag32 f;
    f.lo = *(const uint4*)p0;
    f.hi = *(const uint4*)p1;
    return __builtin_bit_cast(v16bf, f);
}

__device__ __forceinline__ v8f wmma_bf16(v16bf a, v16bf b, v8f c) {
    return __builtin_amdgcn_wmma_f32_16x16x32_bf16(false, a, false, b,
                                                   (short)0, c, false, false);
}

// ---------------------------------------------------------------------------
// Kernel 1: fused Q/K/V projection.  Y = X @ W^T + b  (fp32 in, bf16 out)
// Output layout [B,H,T,DK].  1/sqrt(DK) folded into Q (z==0).
// Block: 256 thr (8 waves).  Tile: 64x64.  grid = (N/64, M/64, 3)
// ---------------------------------------------------------------------------
__global__ __launch_bounds__(256)
void proj_qkv(const float* __restrict__ q_in, const float* __restrict__ k_in,
              const float* __restrict__ v_in,
              const float* __restrict__ Wq, const float* __restrict__ bq,
              const float* __restrict__ Wk, const float* __restrict__ bk,
              const float* __restrict__ Wv, const float* __restrict__ bv,
              unsigned short* __restrict__ qb, unsigned short* __restrict__ kb,
              unsigned short* __restrict__ vb)
{
    const int z = blockIdx.z;
    const float* X    = (z == 0) ? q_in : (z == 1) ? k_in : v_in;
    const float* W    = (z == 0) ? Wq   : (z == 1) ? Wk   : Wv;
    const float* bias = (z == 0) ? bq   : (z == 1) ? bk   : bv;
    unsigned short* out = (z == 0) ? qb : (z == 1) ? kb : vb;
    const float oscale = (z == 0) ? 0.125f : 1.0f;   // 1/sqrt(64) for Q

    __shared__ alignas(16) unsigned short As[64 * 72];
    __shared__ alignas(16) unsigned short Bs[64 * 72];

    const int tid  = threadIdx.x;
    const int lane = tid & 31;
    const int wave = tid >> 5;
    const int rm   = wave >> 1;          // row block 0..3
    const int c0   = (wave & 1) * 2;     // first of two col tiles
    const int lrow = lane & 15;
    const int g    = lane >> 4;
    const int kA   = g * 8;              // A-frag K base per ISA layout
    const int kB   = g * 8;              // B-frag K base per ISA layout

    const int m0 = blockIdx.y * 64;
    const int n0 = blockIdx.x * 64;

    v8f acc0 = {}; v8f acc1 = {};

    for (int kt = 0; kt < DD; kt += 64) {
        __syncthreads();
        // stage A (X rows) and B (W rows == W^T cols), fp32 -> bf16
        for (int t = 0; t < 4; ++t) {
            int idx = tid + t * 256;                 // 0..1023
            int r   = idx >> 4;                      // 0..63
            int c4  = (idx & 15) << 2;               // 0..60 step 4
            float4 a = *(const float4*)(X + (size_t)(m0 + r) * DD + kt + c4);
            unsigned int a0 = f2bf(a.x) | ((unsigned int)f2bf(a.y) << 16);
            unsigned int a1 = f2bf(a.z) | ((unsigned int)f2bf(a.w) << 16);
            *(uint2*)(As + r * 72 + c4) = make_uint2(a0, a1);
            float4 w = *(const float4*)(W + (size_t)(n0 + r) * DD + kt + c4);
            unsigned int b0 = f2bf(w.x) | ((unsigned int)f2bf(w.y) << 16);
            unsigned int b1 = f2bf(w.z) | ((unsigned int)f2bf(w.w) << 16);
            *(uint2*)(Bs + r * 72 + c4) = make_uint2(b0, b1);
        }
        __syncthreads();

        for (int kh = 0; kh < 2; ++kh) {
            const unsigned short* ap = As + (rm * 16 + lrow) * 72 + kh * 32 + kA;
            v16bf af = load_frag(ap, ap + 16);
            const unsigned short* b0p = Bs + (c0 * 16 + lrow) * 72 + kh * 32 + kB;
            const unsigned short* b1p = Bs + ((c0 + 1) * 16 + lrow) * 72 + kh * 32 + kB;
            v16bf bf0 = load_frag(b0p, b0p + 16);
            v16bf bf1 = load_frag(b1p, b1p + 16);
            acc0 = wmma_bf16(af, bf0, acc0);
            acc1 = wmma_bf16(af, bf1, acc1);
        }
    }

    // epilogue: + bias, * scale, bf16, scatter into [B,H,T,DK]
    for (int cc = 0; cc < 2; ++cc) {
        v8f acc = cc ? acc1 : acc0;
        int n  = n0 + (c0 + cc) * 16 + lrow;
        int h  = n >> 6, dk = n & 63;
        float bsv = bias[n];
        for (int r = 0; r < 8; ++r) {
            int m = m0 + rm * 16 + r + 8 * g;
            int b = m >> 10, t = m & 1023;
            float y = (acc[r] + bsv) * oscale;
            out[(((size_t)b * HH + h) * TT + t) * DKK + dk] = f2bf(y);
        }
    }
}

// ---------------------------------------------------------------------------
// Kernel 2: flash attention with concept bias + key mask.
// One block per (h, q-tile of 64, b).  8 waves; wave owns 16 rows x 32 cols.
// S = Q K^T (scale pre-folded), S += 5*qa_concept (h<C), mask -> -1e9,
// online softmax, O += P V.   Q/K frags loaded straight from global bf16.
// ---------------------------------------------------------------------------
__global__ __launch_bounds__(256)
void flash_attn(const unsigned short* __restrict__ qb,
                const unsigned short* __restrict__ kb,
                const unsigned short* __restrict__ vb,
                const float* __restrict__ qa,      // [B,T,T,C]
                const int*   __restrict__ mask,    // [B,T]
                unsigned short* __restrict__ xb)   // [B,T,D] bf16
{
    const int h  = blockIdx.x;
    const int q0 = blockIdx.y * 64;
    const int b  = blockIdx.z;
    const int bh = b * HH + h;

    __shared__ alignas(16) unsigned short Vt[64 * 72]; // V transposed [dk][kv]
    __shared__ alignas(16) unsigned short Ps[64 * 72]; // exp(S-m) bf16
    __shared__ float Ss[64 * 67];                      // raw scores
    __shared__ float fscale[64];
    __shared__ float linv[64];

    const int tid  = threadIdx.x;
    const int lane = tid & 31;
    const int wave = tid >> 5;
    const int rm   = wave >> 1;
    const int c0   = (wave & 1) * 2;
    const int lrow = lane & 15;
    const int g    = lane >> 4;
    const int kA   = g * 8;
    const int kB   = g * 8;

    // Q fragments (fixed for whole block): rows rm*16+lrow, dk split 0..31/32..63
    const unsigned short* qrow =
        qb + ((size_t)bh * TT + q0 + rm * 16 + lrow) * DKK;
    v16bf qf0 = load_frag(qrow + kA,       qrow + kA + 16);
    v16bf qf1 = load_frag(qrow + 32 + kA,  qrow + 32 + kA + 16);

    v8f o0 = {}, o1 = {};
    float m_run = -3.0e38f, l_run = 0.0f;   // live in lanes tid<64

    for (int kv0 = 0; kv0 < TT; kv0 += 64) {
        // prefetch next chunk of the streaming 256MB qa_concept tensor
        if (h < CC && tid < 64)
            __builtin_prefetch(qa + (((size_t)b * TT + q0 + tid) * TT + kv0) * CC, 0, 0);

        // stage V tile transposed into LDS
        for (int t = 0; t < 4; ++t) {
            int idx = tid + t * 256;
            int r   = idx >> 4;                 // kv 0..63
            int c4  = (idx & 15) << 2;          // dk 0..60
            uint2 d = *(const uint2*)(vb + ((size_t)bh * TT + kv0 + r) * DKK + c4);
            Vt[(c4 + 0) * 72 + r] = (unsigned short)(d.x & 0xFFFFu);
            Vt[(c4 + 1) * 72 + r] = (unsigned short)(d.x >> 16);
            Vt[(c4 + 2) * 72 + r] = (unsigned short)(d.y & 0xFFFFu);
            Vt[(c4 + 3) * 72 + r] = (unsigned short)(d.y >> 16);
        }

        // S = Q K^T  (K frags direct from global; B-frag: col=kv, K-dim=dk)
        v8f s0 = {}, s1 = {};
        {
            const unsigned short* k0p =
                kb + ((size_t)bh * TT + kv0 + c0 * 16 + lrow) * DKK;
            const unsigned short* k1p =
                kb + ((size_t)bh * TT + kv0 + (c0 + 1) * 16 + lrow) * DKK;
            v16bf kf00 = load_frag(k0p + kB,      k0p + kB + 16);
            v16bf kf01 = load_frag(k0p + 32 + kB, k0p + 32 + kB + 16);
            v16bf kf10 = load_frag(k1p + kB,      k1p + kB + 16);
            v16bf kf11 = load_frag(k1p + 32 + kB, k1p + 32 + kB + 16);
            s0 = wmma_bf16(qf0, kf00, s0);
            s0 = wmma_bf16(qf1, kf01, s0);
            s1 = wmma_bf16(qf0, kf10, s1);
            s1 = wmma_bf16(qf1, kf11, s1);
        }

        // bias + mask, write raw scores to LDS
        const int kvA = kv0 + c0 * 16 + lrow;
        const int kvB = kv0 + (c0 + 1) * 16 + lrow;
        const int mA = mask[b * TT + kvA];
        const int mB = mask[b * TT + kvB];
        for (int r = 0; r < 8; ++r) {
            int qg = q0 + rm * 16 + r + 8 * g;
            if (h < CC) {
                const float* qap = qa + ((size_t)b * TT + qg) * TT * CC + h;
                s0[r] += LAMBDV * qap[(size_t)kvA * CC];
                s1[r] += LAMBDV * qap[(size_t)kvB * CC];
            }
            if (mA == 0) s0[r] = -1e9f;
            if (mB == 0) s1[r] = -1e9f;
            int row = rm * 16 + r + 8 * g;
            Ss[row * 67 + c0 * 16 + lrow]       = s0[r];
            Ss[row * 67 + (c0 + 1) * 16 + lrow] = s1[r];
        }
        __syncthreads();

        // online softmax: one lane per row
        if (tid < 64) {
            const float* srow = Ss + tid * 67;
            float mx = srow[0];
            for (int c = 1; c < 64; ++c) mx = fmaxf(mx, srow[c]);
            float m_new = fmaxf(m_run, mx);
            float f = __expf(m_run - m_new);
            float lsum = 0.0f;
            unsigned short* prow = Ps + tid * 72;
            for (int c = 0; c < 64; ++c) {
                float p = __expf(srow[c] - m_new);
                lsum += p;
                prow[c] = f2bf(p);
            }
            l_run = l_run * f + lsum;
            m_run = m_new;
            fscale[tid] = f;
        }
        __syncthreads();

        // rescale O, then O += P V
        for (int r = 0; r < 8; ++r) {
            float f = fscale[rm * 16 + r + 8 * g];
            o0[r] *= f; o1[r] *= f;
        }
        {
            const unsigned short* prow = Ps + (rm * 16 + lrow) * 72;
            v16bf pf0 = load_frag(prow + kA,      prow + kA + 16);
            v16bf pf1 = load_frag(prow + 32 + kA, prow + 32 + kA + 16);
            const unsigned short* v0p = Vt + (c0 * 16 + lrow) * 72;
            const unsigned short* v1p = Vt + ((c0 + 1) * 16 + lrow) * 72;
            v16bf vf00 = load_frag(v0p + kB,      v0p + kB + 16);
            v16bf vf01 = load_frag(v0p + 32 + kB, v0p + 32 + kB + 16);
            v16bf vf10 = load_frag(v1p + kB,      v1p + kB + 16);
            v16bf vf11 = load_frag(v1p + 32 + kB, v1p + 32 + kB + 16);
            o0 = wmma_bf16(pf0, vf00, o0);
            o0 = wmma_bf16(pf1, vf01, o0);
            o1 = wmma_bf16(pf0, vf10, o1);
            o1 = wmma_bf16(pf1, vf11, o1);
        }
        __syncthreads();   // protect Vt/Ss/Ps for next iteration
    }

    if (tid < 64) linv[tid] = 1.0f / l_run;
    __syncthreads();

    // normalize + store attention output as bf16 in [B,T,D] (col = h*64 + dk)
    for (int r = 0; r < 8; ++r) {
        int qg  = q0 + rm * 16 + r + 8 * g;
        float inv = linv[rm * 16 + r + 8 * g];
        size_t base = ((size_t)b * TT + qg) * DD + h * DKK + c0 * 16 + lrow;
        xb[base]      = f2bf(o0[r] * inv);
        xb[base + 16] = f2bf(o1[r] * inv);
    }
}

// ---------------------------------------------------------------------------
// Kernel 3: output projection.  out = Xb(bf16) @ Wo^T + bo  -> fp32
// A-tile staging is a pure bf16 copy -> use GLOBAL_LOAD_ASYNC_TO_LDS_B128
// (ASYNCcnt-tracked DMA, no VGPR round-trip) when the toolchain exposes it.
// ---------------------------------------------------------------------------
__global__ __launch_bounds__(256)
void out_proj(const unsigned short* __restrict__ Xb,
              const float* __restrict__ Wo, const float* __restrict__ bo,
              float* __restrict__ out)
{
    __shared__ alignas(16) unsigned short As[64 * 72];
    __shared__ alignas(16) unsigned short Bs[64 * 72];

    const int tid  = threadIdx.x;
    const int lane = tid & 31;
    const int wave = tid >> 5;
    const int rm   = wave >> 1;
    const int c0   = (wave & 1) * 2;
    const int lrow = lane & 15;
    const int g    = lane >> 4;
    const int kA   = g * 8;
    const int kB   = g * 8;

    const int m0 = blockIdx.y * 64;
    const int n0 = blockIdx.x * 64;

    v8f acc0 = {}, acc1 = {};

    for (int kt = 0; kt < DD; kt += 64) {
        __syncthreads();
#if USE_ASYNC_LDS
        // A tile: async global->LDS DMA, 16B per lane, 2 rounds (8KB total)
        for (int t = 0; t < 2; ++t) {
            int idx = tid + t * 256;                 // 0..511
            int r   = idx >> 3;                      // 0..63
            int c8  = (idx & 7) << 3;                // 0..56 step 8
            const unsigned short* gp = Xb + (size_t)(m0 + r) * DD + kt + c8;
            unsigned short* lp = As + r * 72 + c8;
            __builtin_amdgcn_global_load_async_to_lds_b128(
                (as1_v4i*)(uintptr_t)gp,
                (as3_v4i*)(unsigned)(uintptr_t)lp,
                0, 0);
        }
#else
        for (int t = 0; t < 4; ++t) {
            int idx = tid + t * 256;
            int r   = idx >> 4;
            int c4  = (idx & 15) << 2;
            *(uint2*)(As + r * 72 + c4) =
                *(const uint2*)(Xb + (size_t)(m0 + r) * DD + kt + c4);
        }
#endif
        // B tile: fp32 -> bf16 conversion path
        for (int t = 0; t < 4; ++t) {
            int idx = tid + t * 256;
            int r   = idx >> 4;
            int c4  = (idx & 15) << 2;
            float4 w = *(const float4*)(Wo + (size_t)(n0 + r) * DD + kt + c4);
            unsigned int b0 = f2bf(w.x) | ((unsigned int)f2bf(w.y) << 16);
            unsigned int b1 = f2bf(w.z) | ((unsigned int)f2bf(w.w) << 16);
            *(uint2*)(Bs + r * 72 + c4) = make_uint2(b0, b1);
        }
#if USE_ASYNC_LDS
        __builtin_amdgcn_s_wait_asynccnt(0);
#endif
        __syncthreads();

        for (int kh = 0; kh < 2; ++kh) {
            const unsigned short* ap = As + (rm * 16 + lrow) * 72 + kh * 32 + kA;
            v16bf af = load_frag(ap, ap + 16);
            const unsigned short* b0p = Bs + (c0 * 16 + lrow) * 72 + kh * 32 + kB;
            const unsigned short* b1p = Bs + ((c0 + 1) * 16 + lrow) * 72 + kh * 32 + kB;
            acc0 = wmma_bf16(af, load_frag(b0p, b0p + 16), acc0);
            acc1 = wmma_bf16(af, load_frag(b1p, b1p + 16), acc1);
        }
    }

    for (int cc = 0; cc < 2; ++cc) {
        v8f acc = cc ? acc1 : acc0;
        int n = n0 + (c0 + cc) * 16 + lrow;
        float bsv = bo[n];
        for (int r = 0; r < 8; ++r) {
            int m = m0 + rm * 16 + r + 8 * g;
            out[(size_t)m * DD + n] = acc[r] + bsv;
        }
    }
}

// ---------------------------------------------------------------------------
extern "C" void kernel_launch(void* const* d_in, const int* in_sizes, int n_in,
                              void* d_out, int out_size, void* d_ws, size_t ws_size,
                              hipStream_t stream)
{
    const float* qa    = (const float*)d_in[0];
    const float* query = (const float*)d_in[1];
    const float* key   = (const float*)d_in[2];
    const float* value = (const float*)d_in[3];
    const float* Wq = (const float*)d_in[4];
    const float* bq = (const float*)d_in[5];
    const float* Wk = (const float*)d_in[6];
    const float* bk = (const float*)d_in[7];
    const float* Wv = (const float*)d_in[8];
    const float* bv = (const float*)d_in[9];
    const float* Wo = (const float*)d_in[10];
    const float* bo = (const float*)d_in[11];
    const int*   mask = (const int*)d_in[12];
    float* out = (float*)d_out;

    // workspace: qb/kb/vb [B,H,T,DK] bf16 + xb [B,T,D] bf16  = 64 MB total
    const size_t per = (size_t)BB * HH * TT * DKK;   // 8M elements
    unsigned short* qb = (unsigned short*)d_ws;
    unsigned short* kb = qb + per;
    unsigned short* vb = kb + per;
    unsigned short* xb = vb + per;

    dim3 blk(256);
    proj_qkv<<<dim3(DD / 64, (BB * TT) / 64, 3), blk, 0, stream>>>(
        query, key, value, Wq, bq, Wk, bk, Wv, bv, qb, kb, vb);
    flash_attn<<<dim3(HH, TT / 64, BB), blk, 0, stream>>>(
        qb, kb, vb, qa, mask, xb);
    out_proj<<<dim3(DD / 64, (BB * TT) / 64), blk, 0, stream>>>(
        xb, Wo, bo, out);
}
